// TFMultiHeadAttention_5033701671140
// MI455X (gfx1250) — matrix operand order repaired and checked by reading, not compiled, not run on hardware
//
#include <hip/hip_runtime.h>
#include <hip/hip_bf16.h>

// ---------------------------------------------------------------------------
// MI455X (gfx1250) multi-head attention forward.
// B=2, S=2048, D=1024, H=16, DH=64.
// Outputs: out [B,S,D] fp32  followed by  attn [B,H,S,S] fp32 in d_out.
// All matmuls via V_WMMA_F32_16X16X32_BF16 (wave32).
// ---------------------------------------------------------------------------

#define Bx 2
#define Sx 2048
#define Dx 1024
#define Hx 16
#define DHx 64
#define Mx (Bx * Sx)       // 4096 rows for all projection GEMMs

typedef __attribute__((ext_vector_type(16))) __bf16 v16bf;
typedef __attribute__((ext_vector_type(8)))  float  v8f;

union Frag {
    v16bf v;
    unsigned int u[8];
};

// --- A-matrix (16x32 bf16, MxK) fragment from row-major memory -------------
// lane: row m = lane&15 supplied via rowp; half=lane>>4 selects K subrange.
// VGPR j<4 : K = half*8 + 2j ; VGPR j>=4 : K = 16 + half*8 + 2(j-4).
__device__ inline v16bf load_a_frag(const __bf16* __restrict__ rowp, int kk, int lane) {
    Frag f;
    const int h = lane >> 4;
#pragma unroll
    for (int j = 0; j < 8; ++j) {
        const int k = (j < 4) ? (h * 8 + 2 * j) : (16 + h * 8 + 2 * (j - 4));
        f.u[j] = *(const unsigned int*)(rowp + kk + k);
    }
    return f.v;
}

// --- B-matrix (32x16 bf16, KxN) fragment ----------------------------------
// colp points at column n (elements contiguous along K).
// VGPR j<4 : K = half*8 + 2j ; VGPR j>=4 : K = 16 + half*8 + 2(j-4).
__device__ inline v16bf load_b_frag(const __bf16* __restrict__ colp, int kk, int lane) {
    Frag f;
    const int h = lane >> 4;
#pragma unroll
    for (int j = 0; j < 8; ++j) {
        const int k = kk + ((j < 4) ? (h * 8 + 2 * j) : (16 + h * 8 + 2 * (j - 4)));
        f.u[j] = *(const unsigned int*)(colp + k);
    }
    return f.v;
}

__device__ inline v8f wmma_bf16(v16bf a, v16bf b, v8f c) {
    return __builtin_amdgcn_wmma_f32_16x16x32_bf16(false, a, false, b, (short)0, c,
                                                   false, false);
}

// ---------------------------------------------------------------------------
// fp32 -> bf16 (same layout), 4 elements per thread
// ---------------------------------------------------------------------------
__global__ void cvt_bf16_kernel(const float* __restrict__ in, __bf16* __restrict__ out,
                                int n) {
    int i = (blockIdx.x * blockDim.x + threadIdx.x) * 4;
    if (i + 3 < n) {
        float4 v = *(const float4*)(in + i);
        out[i + 0] = (__bf16)v.x;
        out[i + 1] = (__bf16)v.y;
        out[i + 2] = (__bf16)v.z;
        out[i + 3] = (__bf16)v.w;
    }
}

// ---------------------------------------------------------------------------
// fp32 W[K=1024][N=1024] -> bf16 Wt[N][K] (transposed so B-fragments are
// contiguous 32-bit loads along K)
// ---------------------------------------------------------------------------
__global__ void cvt_wt_transpose_kernel(const float* __restrict__ W,
                                        __bf16* __restrict__ Wt) {
    int idx = blockIdx.x * blockDim.x + threadIdx.x;   // 1M elements
    int k = idx >> 10;
    int n = idx & 1023;
    Wt[n * Dx + k] = (__bf16)W[idx];
}

// ---------------------------------------------------------------------------
// GEMM: C[4096x1024] = A[4096x1024](bf16) @ Wt^T(bf16, stored [N][K]) + bias
// grid (64,16), 128 threads (4 waves). Wave w owns n-tile w; loops 4 m-tiles.
// mode 0: bf16 row-major   (Q,K projections)
// mode 1: bf16 V transposed per head: Vt[((b*H+h)*64+dh)*2048 + s]
// mode 2: fp32 row-major to d_out (final output GEMM)
// ---------------------------------------------------------------------------
__global__ void gemm_bias_wmma_kernel(const __bf16* __restrict__ A,
                                      const __bf16* __restrict__ Wt,
                                      const float* __restrict__ bias,
                                      void* __restrict__ out, int mode) {
    const int lane = threadIdx.x & 31;
    const int wid  = threadIdx.x >> 5;
    const int m_blk = blockIdx.x * 64;
    const int ncol  = blockIdx.y * 64 + wid * 16 + (lane & 15);

    const __bf16* bcol = Wt + (size_t)ncol * Dx;

    v8f acc[4] = {};
    for (int kk = 0; kk < Dx; kk += 32) {
        v16bf bfrag = load_b_frag(bcol, kk, lane);
#pragma unroll
        for (int mt = 0; mt < 4; ++mt) {
            const int m = m_blk + mt * 16 + (lane & 15);
            v16bf afrag = load_a_frag(A + (size_t)m * Dx, kk, lane);
            acc[mt] = wmma_bf16(afrag, bfrag, acc[mt]);
        }
    }

    const float bn = bias[ncol];
#pragma unroll
    for (int mt = 0; mt < 4; ++mt) {
        const int mbase = m_blk + mt * 16 + (lane >> 4) * 8;
#pragma unroll
        for (int r = 0; r < 8; ++r) {
            const int m = mbase + r;
            const float v = acc[mt][r] + bn;
            if (mode == 0) {
                ((__bf16*)out)[(size_t)m * Dx + ncol] = (__bf16)v;
            } else if (mode == 1) {
                const int bI = m >> 11, s = m & 2047;
                const int hh = ncol >> 6, dh = ncol & 63;
                ((__bf16*)out)[(size_t)((bI * Hx + hh) * DHx + dh) * Sx + s] = (__bf16)v;
            } else {
                ((float*)out)[(size_t)m * Dx + ncol] = v;
            }
        }
    }
}

// ---------------------------------------------------------------------------
// Attention: per workgroup = one (b,h) and one 16-row q-tile.
// grid (S/16, B*H), 128 threads (4 waves).
// Phase 1: scores = (Q K^T)*scale + mask  -> LDS (16 x 2048 fp32, 128 KB)
// Phase 2: softmax in LDS, write normalized attn (fp32) to d_out
// Phase 3: ctx = attn @ V (bf16 WMMA, A from LDS, B from transposed V)
// ---------------------------------------------------------------------------
__global__ void attn_kernel(const __bf16* __restrict__ Qp,
                            const __bf16* __restrict__ Kp,
                            const __bf16* __restrict__ Vt,
                            const float* __restrict__ mask,
                            float* __restrict__ attn_out,
                            __bf16* __restrict__ ctxb) {
    extern __shared__ float sc[];          // 16*2048 floats
    float* red    = sc + 16 * Sx;          // 128 floats (per-row partials)
    float* rowred = red + 128;             // 16 floats (rowmax, then 1/rowsum)

    const int tid  = threadIdx.x;
    const int lane = tid & 31;
    const int wid  = tid >> 5;
    const int bh = blockIdx.y;
    const int b = bh >> 4, h = bh & 15;
    const int q0 = blockIdx.x * 16;

    // ---- Q fragments (shared by all key tiles this wave processes) ----
    const int qrow = b * Sx + q0 + (lane & 15);
    const __bf16* qptr = Qp + (size_t)qrow * Dx + h * DHx;
    const v16bf qa0 = load_a_frag(qptr, 0, lane);
    const v16bf qa1 = load_a_frag(qptr, 32, lane);

    // ---- Phase 1: scores -> LDS ----
    const float* mrow = mask + (size_t)b * Sx * Sx;
    for (int t = wid; t < Sx / 16; t += 4) {
        const int col0 = t * 16;
        const int key  = b * Sx + col0 + (lane & 15);
        const __bf16* kcol = Kp + (size_t)key * Dx + h * DHx;   // elems along dh
        v16bf kf0 = load_b_frag(kcol, 0, lane);
        v16bf kf1 = load_b_frag(kcol, 32, lane);
        v8f acc = {};
        acc = wmma_bf16(qa0, kf0, acc);
        acc = wmma_bf16(qa1, kf1, acc);
        const int n = lane & 15;
#pragma unroll
        for (int r = 0; r < 8; ++r) {
            const int m = (lane >> 4) * 8 + r;
            const float v = acc[r] * 0.125f +
                            mrow[(size_t)(q0 + m) * Sx + col0 + n];
            sc[m * Sx + col0 + n] = v;
        }
    }
    __syncthreads();

    // ---- Phase 2: softmax over each of 16 rows (8 threads per row) ----
    const int row = tid >> 3, sub = tid & 7;
    float lm = -3.4e38f;
    for (int c = sub; c < Sx; c += 8) lm = fmaxf(lm, sc[row * Sx + c]);
    red[row * 8 + sub] = lm;
    __syncthreads();
    if (sub == 0) {
        float m = red[row * 8];
#pragma unroll
        for (int i = 1; i < 8; ++i) m = fmaxf(m, red[row * 8 + i]);
        rowred[row] = m;
    }
    __syncthreads();
    const float rm = rowred[row];
    float ls = 0.f;
    for (int c = sub; c < Sx; c += 8) {
        const float e = __expf(sc[row * Sx + c] - rm);
        sc[row * Sx + c] = e;
        ls += e;
    }
    red[row * 8 + sub] = ls;
    __syncthreads();
    if (sub == 0) {
        float s = 0.f;
#pragma unroll
        for (int i = 0; i < 8; ++i) s += red[row * 8 + i];
        rowred[row] = 1.0f / s;
    }
    __syncthreads();

    // normalize in LDS + coalesced fp32 attn write
    const size_t abase = ((size_t)bh * Sx + q0) * Sx;
    for (int idx = tid; idx < 16 * Sx; idx += 128) {
        const int r = idx >> 11;
        const float v = sc[idx] * rowred[r];
        sc[idx] = v;
        attn_out[abase + idx] = v;
    }
    __syncthreads();

    // ---- Phase 3: ctx[16 x 64] = attn[16 x 2048] @ V[2048 x 64] ----
    // wave w owns dh columns [w*16, w*16+16)
    const int n0 = wid * 16;
    const int vrow = (b * Hx + h) * DHx + n0 + (lane & 15);
    const __bf16* vcol = Vt + (size_t)vrow * Sx;   // elems contiguous along s
    v8f acc = {};
    for (int kk = 0; kk < Sx; kk += 32) {
        Frag fa;
        const int m  = lane & 15;
        const int hh = lane >> 4;
#pragma unroll
        for (int j = 0; j < 8; ++j) {
            const int k = kk + ((j < 4) ? (hh * 8 + 2 * j) : (16 + hh * 8 + 2 * (j - 4)));
            fa.v[2 * j]     = (__bf16)sc[m * Sx + k];
            fa.v[2 * j + 1] = (__bf16)sc[m * Sx + k + 1];
        }
        v16bf bf = load_b_frag(vcol, kk, lane);
        acc = wmma_bf16(fa.v, bf, acc);
    }
    const int n = lane & 15;
#pragma unroll
    for (int r = 0; r < 8; ++r) {
        const int m = q0 + (lane >> 4) * 8 + r;
        ctxb[((size_t)(b * Sx + m)) * Dx + h * DHx + n0 + n] = (__bf16)acc[r];
    }
}

// ---------------------------------------------------------------------------
extern "C" void kernel_launch(void* const* d_in, const int* in_sizes, int n_in,
                              void* d_out, int out_size, void* d_ws, size_t ws_size,
                              hipStream_t stream) {
    (void)in_sizes; (void)n_in; (void)out_size; (void)ws_size;

    const float* q_in = (const float*)d_in[0];
    const float* k_in = (const float*)d_in[1];
    const float* v_in = (const float*)d_in[2];
    const float* mask = (const float*)d_in[3];
    const float* Wq = (const float*)d_in[4];  const float* bq = (const float*)d_in[5];
    const float* Wk = (const float*)d_in[6];  const float* bk = (const float*)d_in[7];
    const float* Wv = (const float*)d_in[8];  const float* bv = (const float*)d_in[9];
    const float* Wo = (const float*)d_in[10]; const float* bo = (const float*)d_in[11];

    float* out_ptr  = (float*)d_out;                         // [B,S,D]
    float* attn_ptr = out_ptr + (size_t)Bx * Sx * Dx;        // [B,H,S,S]

    // workspace carve-up (bf16 buffers)
    const size_t SZ_ACT = (size_t)Mx * Dx * sizeof(__bf16);  // 8 MB
    const size_t SZ_W   = (size_t)Dx * Dx * sizeof(__bf16);  // 2 MB
    char* p = (char*)d_ws;
    __bf16* qb   = (__bf16*)p; p += SZ_ACT;
    __bf16* kb   = (__bf16*)p; p += SZ_ACT;
    __bf16* vb   = (__bf16*)p; p += SZ_ACT;
    __bf16* WqT  = (__bf16*)p; p += SZ_W;
    __bf16* WkT  = (__bf16*)p; p += SZ_W;
    __bf16* WvT  = (__bf16*)p; p += SZ_W;
    __bf16* WoT  = (__bf16*)p; p += SZ_W;
    __bf16* Qp   = (__bf16*)p; p += SZ_ACT;
    __bf16* Kp   = (__bf16*)p; p += SZ_ACT;
    __bf16* VtT  = (__bf16*)p; p += SZ_ACT;   // [B*H*DH, S]
    __bf16* ctxb = (__bf16*)p; p += SZ_ACT;

    const int nAct = Mx * Dx;                 // 4,194,304
    // fp32 -> bf16 activations
    cvt_bf16_kernel<<<nAct / (256 * 4), 256, 0, stream>>>(q_in, qb, nAct);
    cvt_bf16_kernel<<<nAct / (256 * 4), 256, 0, stream>>>(k_in, kb, nAct);
    cvt_bf16_kernel<<<nAct / (256 * 4), 256, 0, stream>>>(v_in, vb, nAct);
    // fp32 -> bf16 transposed weights
    const int nW = Dx * Dx;                   // 1,048,576
    cvt_wt_transpose_kernel<<<nW / 256, 256, 0, stream>>>(Wq, WqT);
    cvt_wt_transpose_kernel<<<nW / 256, 256, 0, stream>>>(Wk, WkT);
    cvt_wt_transpose_kernel<<<nW / 256, 256, 0, stream>>>(Wv, WvT);
    cvt_wt_transpose_kernel<<<nW / 256, 256, 0, stream>>>(Wo, WoT);

    dim3 ggrid(Mx / 64, Dx / 64);             // (64, 16)
    gemm_bias_wmma_kernel<<<ggrid, 128, 0, stream>>>(qb, WqT, bq, Qp, 0);
    gemm_bias_wmma_kernel<<<ggrid, 128, 0, stream>>>(kb, WkT, bk, Kp, 0);
    gemm_bias_wmma_kernel<<<ggrid, 128, 0, stream>>>(vb, WvT, bv, VtT, 1);

    dim3 agrid(Sx / 16, Bx * Hx);             // (128, 32)
    const size_t lds_bytes = (size_t)16 * Sx * sizeof(float) + (128 + 16) * sizeof(float);
    attn_kernel<<<agrid, 128, lds_bytes, stream>>>(Qp, Kp, VtT, mask, attn_ptr, ctxb);

    gemm_bias_wmma_kernel<<<ggrid, 128, 0, stream>>>(ctxb, WoT, bo, out_ptr, 2);
}
